// SelfAttention_62088047231620
// MI455X (gfx1250) — matrix operand order repaired and checked by reading, not compiled, not run on hardware
//
#include <hip/hip_runtime.h>

#define EMB   1024
#define HEADS 16
#define HSZ   64
#define BB    4
#define TT    1024

typedef __attribute__((ext_vector_type(16))) __bf16          v16bf;
typedef __attribute__((ext_vector_type(8)))  float           v8f;
typedef __attribute__((ext_vector_type(16))) unsigned short  v16u;
typedef __attribute__((ext_vector_type(8)))  unsigned short  v8u;

union BF16Frag {
    v16bf          v;
    v16u           u;
    v8u            h[2];
    unsigned short s[16];
};

// ---- CDNA5 async global->LDS support (guarded; fallback = load + ds_store) --
#if defined(__has_builtin)
#if __has_builtin(__builtin_amdgcn_global_load_async_to_lds_b128)
#define HAVE_ASYNC_LDS 1
#endif
#endif

// Builtin signature (from hipcc diagnostic): param1 = AS1 pointer to
// 'int __attribute__((vector_size(16)))', param2 = AS3 pointer to same.
typedef int v4i_ __attribute__((vector_size(16)));
#define GP128(p) ((__attribute__((address_space(1))) v4i_*)(p))
#define LP128(p) ((__attribute__((address_space(3))) v4i_*)(p))

__device__ __forceinline__ void wait_async0() {
#if defined(__has_builtin) && __has_builtin(__builtin_amdgcn_s_wait_asynccnt)
    __builtin_amdgcn_s_wait_asynccnt(0);
#else
    asm volatile("s_wait_asynccnt 0x0" ::: "memory");
#endif
}

// float -> bf16, round-to-nearest-even
__device__ __forceinline__ unsigned short f2bf(float f) {
    unsigned int u = __float_as_uint(f);
    unsigned int r = u + 0x7fffu + ((u >> 16) & 1u);
    return (unsigned short)(r >> 16);
}

__device__ __forceinline__ v8f wmma_bf16(v16bf a, v16bf b, v8f c) {
    return __builtin_amdgcn_wmma_f32_16x16x32_bf16(false, a, false, b, (short)0, c,
                                                   false, false);
}

// ---------------------------------------------------------------------------
// f32 -> bf16 cast (4 elements / thread)
// ---------------------------------------------------------------------------
__global__ void cast_f32_to_bf16(const float* __restrict__ in,
                                 unsigned short* __restrict__ out, int n) {
    int i = (blockIdx.x * blockDim.x + threadIdx.x) * 4;
    if (i + 3 < n) {
        float4 f = *(const float4*)(in + i);
        out[i + 0] = f2bf(f.x);
        out[i + 1] = f2bf(f.y);
        out[i + 2] = f2bf(f.z);
        out[i + 3] = f2bf(f.w);
    }
}

// ---------------------------------------------------------------------------
// Y = X * W^T   (X: MxK bf16, W: NxK bf16 row-major)
// 32x32 register tile per wave (2 A-frags x 2 B-frags -> 4 WMMAs / k-step),
// 4 waves / block arranged 2x2 over a 64x64 macro-tile.
// A-fragment: lane m=lane&15, one contiguous 16-elem run at k = half*16
// B-fragment: lane n=lane&15, one contiguous 16-elem run at k = half*16
// Output: bf16 (out16, *scale) or f32 (out32, *scale + bias)
// ---------------------------------------------------------------------------
__global__ __launch_bounds__(128)
void gemm_xWT_wmma(const unsigned short* __restrict__ X,
                   const unsigned short* __restrict__ W,
                   unsigned short* __restrict__ out16,
                   float* __restrict__ out32,
                   const float* __restrict__ bias,
                   float scale, int M, int N, int K) {
    const int lane = threadIdx.x & 31;
    const int wave = threadIdx.x >> 5;
    const int m0   = blockIdx.y * 64 + (wave >> 1) * 32;
    const int n0   = blockIdx.x * 64 + (wave & 1) * 32;

    const int mrow = lane & 15;
    const int half = lane >> 4;

    const unsigned short* arow0 = X + (size_t)(m0 + mrow) * K + half * 16;
    const unsigned short* arow1 = arow0 + (size_t)16 * K;
    const unsigned short* brow0 = W + (size_t)(n0 + mrow) * K + half * 16;
    const unsigned short* brow1 = brow0 + (size_t)16 * K;

    v8f acc00 = {}, acc01 = {}, acc10 = {}, acc11 = {};
#pragma unroll 2
    for (int k0 = 0; k0 < K; k0 += 32) {
        BF16Frag a0, a1, b0, b1;
        a0.h[0] = *(const v8u*)(arow0 + k0);
        a0.h[1] = *(const v8u*)(arow0 + k0 + 8);
        a1.h[0] = *(const v8u*)(arow1 + k0);
        a1.h[1] = *(const v8u*)(arow1 + k0 + 8);
        b0.u    = *(const v16u*)(brow0 + k0);
        b1.u    = *(const v16u*)(brow1 + k0);
        acc00 = wmma_bf16(a0.v, b0.v, acc00);
        acc01 = wmma_bf16(a0.v, b1.v, acc01);
        acc10 = wmma_bf16(a1.v, b0.v, acc10);
        acc11 = wmma_bf16(a1.v, b1.v, acc11);
    }

    const int nc0 = n0 + mrow;
    const int nc1 = n0 + 16 + mrow;
    if (out32) {
        const float bv0 = bias ? bias[nc0] : 0.0f;
        const float bv1 = bias ? bias[nc1] : 0.0f;
#pragma unroll
        for (int i = 0; i < 8; ++i) {
            const size_t r0 = (size_t)(m0 + half * 8 + i) * N;
            const size_t r1 = (size_t)(m0 + 16 + half * 8 + i) * N;
            out32[r0 + nc0] = acc00[i] * scale + bv0;
            out32[r0 + nc1] = acc01[i] * scale + bv1;
            out32[r1 + nc0] = acc10[i] * scale + bv0;
            out32[r1 + nc1] = acc11[i] * scale + bv1;
        }
    } else {
#pragma unroll
        for (int i = 0; i < 8; ++i) {
            const size_t r0 = (size_t)(m0 + half * 8 + i) * N;
            const size_t r1 = (size_t)(m0 + 16 + half * 8 + i) * N;
            out16[r0 + nc0] = f2bf(acc00[i] * scale);
            out16[r0 + nc1] = f2bf(acc01[i] * scale);
            out16[r1 + nc0] = f2bf(acc10[i] * scale);
            out16[r1 + nc1] = f2bf(acc11[i] * scale);
        }
    }
}

// ---------------------------------------------------------------------------
// Causal attention core: one block (128 threads, 4 waves) per (b, h, 16-query
// tile).  Q/K already scaled by EMB^(-1/4) at projection time.
//   Phase 1: S = Q K^T  (waves stride over key tiles; 2 WMMAs per 16x16 tile)
//   Phase 2: causal softmax (8 threads per row, LDS reduction), P -> LDS bf16
//   Phase 3: O = P V ; per 32-key chunk, stage V (32x64 bf16 = 4KB) into LDS
//            via async global->LDS (ASYNCcnt) and gather B-frag columns from
//            LDS.  Wave w owns output head cols [w*16, w*16+16).
// LDS: 16x1024 f32 + 16x1024 bf16 + 16x8 f32 + 32x64 bf16 ~= 101 KB
// ---------------------------------------------------------------------------
__global__ __launch_bounds__(128)
void attn_causal_wmma(const unsigned short* __restrict__ Qb,
                      const unsigned short* __restrict__ Kb,
                      const unsigned short* __restrict__ Vb,
                      unsigned short* __restrict__ Ctx) {
    extern __shared__ char smem[];
    float (*sS)[TT]          = (float (*)[TT])smem;                          // 64 KB
    unsigned short (*sP)[TT] = (unsigned short (*)[TT])(smem + 16 * TT * 4); // 32 KB
    float (*sRed)[8]         = (float (*)[8])(smem + 16 * TT * 6);           // 512 B
    unsigned short* sV       = (unsigned short*)(smem + 16 * TT * 6 + 512);  // 4 KB

    const int QT  = TT / 16;
    const int qt  = blockIdx.x % QT;
    const int h   = (blockIdx.x / QT) % HEADS;
    const int b   = blockIdx.x / (QT * HEADS);
    const int q0  = qt * 16;
    const int nkt = qt + 1;   // valid 16-wide key tiles (causal)

    const int tid  = threadIdx.x;
    const int lane = tid & 31;
    const int wave = tid >> 5;
    const int mrow = lane & 15;
    const int half = lane >> 4;

    // ---- Phase 1: scores -> LDS ----
    const size_t qbase = ((size_t)(b * TT + q0 + mrow)) * EMB + h * HSZ;
    BF16Frag a0, a1;
    a0.h[0] = *(const v8u*)(Qb + qbase + half * 16);
    a0.h[1] = *(const v8u*)(Qb + qbase + half * 16 + 8);
    a1.h[0] = *(const v8u*)(Qb + qbase + 32 + half * 16);
    a1.h[1] = *(const v8u*)(Qb + qbase + 32 + half * 16 + 8);

    for (int kt = wave; kt < nkt; kt += 4) {
        const size_t kbase =
            ((size_t)(b * TT + kt * 16 + mrow)) * EMB + h * HSZ + half * 16;
        BF16Frag b0, b1;
        b0.u = *(const v16u*)(Kb + kbase);
        b1.u = *(const v16u*)(Kb + kbase + 32);
        v8f acc = {};
        acc = wmma_bf16(a0.v, b0.v, acc);
        acc = wmma_bf16(a1.v, b1.v, acc);
#pragma unroll
        for (int i = 0; i < 8; ++i)
            sS[half * 8 + i][kt * 16 + mrow] = acc[i];
    }
    __syncthreads();

    // ---- Phase 2: causal softmax (8 threads per row) ----
    const int r     = tid >> 3;
    const int sub   = tid & 7;
    const int ncols = q0 + r + 1;

    float mx = -1e30f;
    for (int c = sub; c < ncols; c += 8) mx = fmaxf(mx, sS[r][c]);
    sRed[r][sub] = mx;
    __syncthreads();
    float rowmax = sRed[r][0];
#pragma unroll
    for (int j = 1; j < 8; ++j) rowmax = fmaxf(rowmax, sRed[r][j]);
    __syncthreads();

    float sm = 0.0f;
    for (int c = sub; c < ncols; c += 8) sm += __expf(sS[r][c] - rowmax);
    sRed[r][sub] = sm;
    __syncthreads();
    float rowsum = 0.0f;
#pragma unroll
    for (int j = 0; j < 8; ++j) rowsum += sRed[r][j];
    const float inv = 1.0f / rowsum;

    const int ncpad = (nkt * 16 + 31) & ~31;
    for (int c = sub; c < ncpad; c += 8) {
        float p = (c < ncols) ? __expf(sS[r][c] - rowmax) * inv : 0.0f;
        sP[r][c] = f2bf(p);
    }
    __syncthreads();

    // ---- Phase 3: O = P V with async-staged V chunks ----
    v8f acc = {};
    for (int kk = 0; kk < nkt * 16; kk += 32) {
        // Stage V[kk..kk+31][h*64..h*64+63] (4 KB) into sV: 256 x 16B chunks,
        // 2 per thread.  Chunk ch: row = ch>>3, col8 = (ch&7)*8 (u16 units).
#pragma unroll
        for (int q = 0; q < 2; ++q) {
            const int ch   = tid + q * 128;
            const int vrow = ch >> 3;
            const int c8   = (ch & 7) * 8;
            const unsigned short* src =
                Vb + ((size_t)(b * TT + kk + vrow)) * EMB + h * HSZ + c8;
            unsigned short* dst = sV + vrow * HSZ + c8;
#if defined(HAVE_ASYNC_LDS)
            __builtin_amdgcn_global_load_async_to_lds_b128(GP128(src), LP128(dst),
                                                           0, 0);
#else
            *(v8u*)dst = *(const v8u*)src;
#endif
        }
#if defined(HAVE_ASYNC_LDS)
        wait_async0();
#endif
        __syncthreads();

        BF16Frag pa, vb;
        pa.h[0] = *(const v8u*)(&sP[mrow][kk + half * 16]);
        pa.h[1] = *(const v8u*)(&sP[mrow][kk + half * 16 + 8]);
        const int scol = wave * 16 + mrow;   // head col owned by this lane
#pragma unroll
        for (int j = 0; j < 16; ++j)
            vb.s[j] = sV[(half * 16 + j) * HSZ + scol];
        acc = wmma_bf16(pa.v, vb.v, acc);
        __syncthreads();   // sV consumed; safe to overwrite next iteration
    }
#pragma unroll
    for (int i = 0; i < 8; ++i) {
        const int m = half * 8 + i;
        Ctx[((size_t)(b * TT + q0 + m)) * EMB + h * HSZ + wave * 16 + mrow] =
            f2bf(acc[i]);
    }
}

// ---------------------------------------------------------------------------
// Launcher
// ---------------------------------------------------------------------------
extern "C" void kernel_launch(void* const* d_in, const int* in_sizes, int n_in,
                              void* d_out, int out_size, void* d_ws, size_t ws_size,
                              hipStream_t stream) {
    const float* x  = (const float*)d_in[0];
    const float* Wq = (const float*)d_in[1];
    const float* Wk = (const float*)d_in[2];
    const float* Wv = (const float*)d_in[3];
    const float* Wo = (const float*)d_in[4];
    const float* bo = (const float*)d_in[5];
    float* out      = (float*)d_out;

    const size_t NX = (size_t)BB * TT * EMB;
    const size_t NW = (size_t)EMB * EMB;

    unsigned short* xb  = (unsigned short*)d_ws;
    unsigned short* wqb = xb  + NX;
    unsigned short* wkb = wqb + NW;
    unsigned short* wvb = wkb + NW;
    unsigned short* wob = wvb + NW;
    unsigned short* Qb  = wob + NW;
    unsigned short* Kb  = Qb  + NX;
    unsigned short* Vb  = Kb  + NX;
    unsigned short* Ctx = Vb  + NX;

    const float inv4 = 0.17677669529663687f;   // 1 / 1024^(1/4)

    cast_f32_to_bf16<<<dim3((unsigned)(NX / 4 / 256)), 256, 0, stream>>>(x,  xb,  (int)NX);
    cast_f32_to_bf16<<<dim3((unsigned)(NW / 4 / 256)), 256, 0, stream>>>(Wq, wqb, (int)NW);
    cast_f32_to_bf16<<<dim3((unsigned)(NW / 4 / 256)), 256, 0, stream>>>(Wk, wkb, (int)NW);
    cast_f32_to_bf16<<<dim3((unsigned)(NW / 4 / 256)), 256, 0, stream>>>(Wv, wvb, (int)NW);
    cast_f32_to_bf16<<<dim3((unsigned)(NW / 4 / 256)), 256, 0, stream>>>(Wo, wob, (int)NW);

    const int M = BB * TT, N = EMB, K = EMB;
    dim3 ggrid(N / 64, M / 64);
    gemm_xWT_wmma<<<ggrid, 128, 0, stream>>>(xb, wqb, Qb, nullptr, nullptr, inv4, M, N, K);
    gemm_xWT_wmma<<<ggrid, 128, 0, stream>>>(xb, wkb, Kb, nullptr, nullptr, inv4, M, N, K);
    gemm_xWT_wmma<<<ggrid, 128, 0, stream>>>(xb, wvb, Vb, nullptr, nullptr, 1.0f, M, N, K);

    const unsigned nblk = BB * HEADS * (TT / 16);
    const size_t smem   = (size_t)16 * TT * 6 + 512 + (size_t)32 * HSZ * 2;
    attn_causal_wmma<<<nblk, 128, smem, stream>>>(Qb, Kb, Vb, Ctx);

    gemm_xWT_wmma<<<ggrid, 128, 0, stream>>>(Ctx, wob, nullptr, out, bo, 1.0f, M, N, K);
}
